// SwinTransformerBlock_59639915872522
// MI455X (gfx1250) — matrix operand order repaired and checked by reading, not compiled, not run on hardware
//
#include <hip/hip_runtime.h>
#include <hip/hip_bf16.h>
#include <math.h>

// ---------------- problem constants ----------------
#define BATCH 8
#define CDIM  96
#define IMG   224
#define HWSZ  (IMG*IMG)          // 50176
#define NT    (BATCH*HWSZ)       // 401408 tokens, divisible by 16
#define NWIN  (BATCH*32*32)      // 8192 windows
#define NPW   49                 // tokens per 7x7 window
#define QKVD  288
#define HID   384
#define NHEAD 3
#define HD    32

typedef __attribute__((ext_vector_type(16))) _Float16 v16h;
typedef __attribute__((ext_vector_type(8)))  float    v8f;
typedef __attribute__((ext_vector_type(4)))  int      v4i;

#define AS_GLOBAL __attribute__((address_space(1)))
#define AS_LDS    __attribute__((address_space(3)))

#if defined(__has_builtin)
#if __has_builtin(__builtin_amdgcn_global_load_async_to_lds_b128)
#define HAVE_ASYNC_LDS 1
#endif
#endif

// ---------------- WMMA helpers (gfx1250, wave32) ----------------
// A fragment: 16(M) x 32(K) tile of a row-major f16 matrix.
// Layout: lanes 0-15 -> M=lane, K = {0..15}; lanes 16-31 -> M=lane-16, K={16..31}
__device__ __forceinline__ v16h frag_a(const _Float16* base, int lda, int lane) {
  const int m  = lane & 15;
  const int kb = (lane >> 4) << 4;          // 0 or 16
  const _Float16* p = base + m * lda + kb;
  v16h a;
#pragma unroll
  for (int e = 0; e < 8; ++e) a[e] = p[e];
#pragma unroll
  for (int e = 0; e < 8; ++e) a[e + 8] = p[e + 8];
  return a;
}

// B fragment 32(K) x 16(N), source stored as W[N][K] row-major (i.e. B = W^T).
__device__ __forceinline__ v16h frag_b_t(const _Float16* base, int ldn, int lane) {
  const int n  = lane & 15;
  const int kb = (lane >> 4) << 4;          // 0 or 16
  const _Float16* p = base + n * ldn + kb;
  v16h b;
#pragma unroll
  for (int e = 0; e < 16; ++e) b[e] = p[e];
  return b;
}

// B fragment 32(K) x 16(N), source stored K-major: M[k][n] with row stride ldk.
__device__ __forceinline__ v16h frag_b_k(const _Float16* base, int ldk, int lane) {
  const int n  = lane & 15;
  const int kb = (lane >> 4) << 4;          // 0 or 16
  const _Float16* p = base + (size_t)kb * ldk + n;
  v16h b;
#pragma unroll
  for (int e = 0; e < 16; ++e) b[e] = p[(size_t)e * ldk];
  return b;
}

__device__ __forceinline__ v8f wmma16(v16h a, v16h b, v8f c) {
  // D = A x B + C, f32 accumulate  -> v_wmma_f32_16x16x32_f16
  return __builtin_amdgcn_wmma_f32_16x16x32_f16(false, a, false, b, (short)0, c, false, false);
}

// ---------------- utility kernels ----------------
__global__ void cvt_f16_kernel(const float* __restrict__ in, _Float16* __restrict__ out, int n) {
  int i = blockIdx.x * blockDim.x + threadIdx.x;
  if (i < n) out[i] = (_Float16)in[i];
}

// LN1 over channels + cyclic shift(-3,-3) + window partition -> f16 [NT, 96] (window order)
__global__ __launch_bounds__(256) void ln1_shift_part_kernel(
    const float* __restrict__ x, const float* __restrict__ g, const float* __restrict__ bta,
    _Float16* __restrict__ xwin) {
  int row = blockIdx.x * blockDim.x + threadIdx.x;
  if (row >= NT) return;
  int win = row / NPW, t = row - win * NPW;
  int b  = win >> 10;
  int wi = win & 1023;
  int wh = wi >> 5, ww = wi & 31;
  int hs = wh * 7 + t / 7, wsp = ww * 7 + t % 7;     // coords in shifted image
  int hh = hs + 3;  if (hh >= IMG) hh -= IMG;        // shifted[h] = x[(h+3)%224]
  int wp = wsp + 3; if (wp >= IMG) wp -= IMG;
  const float* px = x + (size_t)b * CDIM * HWSZ + (size_t)hh * IMG + wp;
  float s = 0.f, s2 = 0.f;
  for (int c = 0; c < CDIM; ++c) {
    float v = px[(size_t)c * HWSZ];
    s += v; s2 += v * v;
  }
  float mean = s * (1.f / CDIM);
  float var  = s2 * (1.f / CDIM) - mean * mean;
  float inv  = rsqrtf(var + 1e-5f);
  _Float16* po = xwin + (size_t)row * CDIM;
  for (int c = 0; c < CDIM; ++c) {
    float v = px[(size_t)c * HWSZ];
    po[c] = (_Float16)((v - mean) * inv * g[c] + bta[c]);
  }
}

// LN2 over channels on y [B,C,H,W] -> f16 [NT, 96] (raster token order)
__global__ __launch_bounds__(256) void ln2_kernel(
    const float* __restrict__ y, const float* __restrict__ g, const float* __restrict__ bta,
    _Float16* __restrict__ h16) {
  int row = blockIdx.x * blockDim.x + threadIdx.x;
  if (row >= NT) return;
  int b = row / HWSZ, hw = row - b * HWSZ;
  const float* py = y + (size_t)b * CDIM * HWSZ + hw;
  float s = 0.f, s2 = 0.f;
  for (int c = 0; c < CDIM; ++c) {
    float v = py[(size_t)c * HWSZ];
    s += v; s2 += v * v;
  }
  float mean = s * (1.f / CDIM);
  float var  = s2 * (1.f / CDIM) - mean * mean;
  float inv  = rsqrtf(var + 1e-5f);
  _Float16* po = h16 + (size_t)row * CDIM;
  for (int c = 0; c < CDIM; ++c) {
    float v = py[(size_t)c * HWSZ];
    po[c] = (_Float16)((v - mean) * inv * g[c] + bta[c]);
  }
}

// ---------------- GEMM kernels ----------------
// Each wave: 1 M-tile x 6 N-tiles; A fragment loaded once per k-step, reused 6x.

// qkv = xwin @ qkv_w^T + qkv_b : M=NT, N=288 (3 groups of 96), K=96 -> f16 out
__global__ __launch_bounds__(256) void qkv_gemm_kernel(
    const _Float16* __restrict__ A, const _Float16* __restrict__ W,
    const float* __restrict__ bias, _Float16* __restrict__ out) {
  const int lane  = threadIdx.x & 31;
  const int group = blockIdx.x * 8 + (threadIdx.x >> 5);
  const int gm = group / 3, gn = group - gm * 3;
  const int m0 = gm << 4, n0b = gn * 96;
  __builtin_prefetch(W + (size_t)n0b * CDIM, 0, 1);
  v8f acc[6] = {};
#pragma unroll
  for (int k0 = 0; k0 < CDIM; k0 += 32) {
    v16h a = frag_a(A + (size_t)m0 * CDIM + k0, CDIM, lane);
#pragma unroll
    for (int s = 0; s < 6; ++s) {
      v16h b = frag_b_t(W + (size_t)(n0b + 16 * s) * CDIM + k0, CDIM, lane);
      acc[s] = wmma16(a, b, acc[s]);
    }
  }
  const int mh = (lane >> 4) << 3;
#pragma unroll
  for (int s = 0; s < 6; ++s) {
    const int n = n0b + 16 * s + (lane & 15);
    const float bn = bias[n];
#pragma unroll
    for (int r = 0; r < 8; ++r)
      out[(size_t)(m0 + mh + r) * QKVD + n] = (_Float16)(acc[s][r] + bn);
  }
}

// proj GEMM + window-reverse + un-shift + residual: y = x + reverse(ao @ proj_w^T + b)
// M=NT, N=96 (6 tiles, all in one wave), K=96
__global__ __launch_bounds__(256) void proj_gemm_scatter_kernel(
    const _Float16* __restrict__ A, const _Float16* __restrict__ W,
    const float* __restrict__ bias, const float* __restrict__ x, float* __restrict__ y) {
  const int lane = threadIdx.x & 31;
  const int gm   = blockIdx.x * 8 + (threadIdx.x >> 5);
  const int m0   = gm << 4;
  __builtin_prefetch(W, 0, 1);
  v8f acc[6] = {};
#pragma unroll
  for (int k0 = 0; k0 < CDIM; k0 += 32) {
    v16h a = frag_a(A + (size_t)m0 * CDIM + k0, CDIM, lane);
#pragma unroll
    for (int s = 0; s < 6; ++s) {
      v16h b = frag_b_t(W + (size_t)(16 * s) * CDIM + k0, CDIM, lane);
      acc[s] = wmma16(a, b, acc[s]);
    }
  }
  const int mh = (lane >> 4) << 3;
  const int nl = lane & 15;
#pragma unroll
  for (int r = 0; r < 8; ++r) {
    int row = m0 + mh + r;                  // window-order token
    int win = row / NPW, t = row - win * NPW;
    int b_  = win >> 10;
    int wi  = win & 1023;
    int wh  = wi >> 5, ww = wi & 31;
    int hs  = wh * 7 + t / 7, wsp = ww * 7 + t % 7;
    int hh  = hs + 3;  if (hh >= IMG) hh -= IMG;   // reverse of the -3 shift
    int wp  = wsp + 3; if (wp >= IMG) wp -= IMG;
    size_t base0 = (size_t)b_ * CDIM * HWSZ + (size_t)hh * IMG + wp;
#pragma unroll
    for (int s = 0; s < 6; ++s) {
      int n = 16 * s + nl;                  // channel
      size_t idx = base0 + (size_t)n * HWSZ;
      y[idx] = x[idx] + (acc[s][r] + bias[n]);
    }
  }
}

// FC1: h16 @ w1^T + b1, exact GELU -> f16. M=NT, N=384 (4 groups of 96), K=96
__global__ __launch_bounds__(256) void fc1_gemm_kernel(
    const _Float16* __restrict__ A, const _Float16* __restrict__ W,
    const float* __restrict__ bias, _Float16* __restrict__ out) {
  const int lane  = threadIdx.x & 31;
  const int group = blockIdx.x * 8 + (threadIdx.x >> 5);
  const int gm = group >> 2, gn = group & 3;
  const int m0 = gm << 4, n0b = gn * 96;
  __builtin_prefetch(W + (size_t)n0b * CDIM, 0, 1);
  v8f acc[6] = {};
#pragma unroll
  for (int k0 = 0; k0 < CDIM; k0 += 32) {
    v16h a = frag_a(A + (size_t)m0 * CDIM + k0, CDIM, lane);
#pragma unroll
    for (int s = 0; s < 6; ++s) {
      v16h b = frag_b_t(W + (size_t)(n0b + 16 * s) * CDIM + k0, CDIM, lane);
      acc[s] = wmma16(a, b, acc[s]);
    }
  }
  const int mh = (lane >> 4) << 3;
#pragma unroll
  for (int s = 0; s < 6; ++s) {
    const int n = n0b + 16 * s + (lane & 15);
    const float bn = bias[n];
#pragma unroll
    for (int r = 0; r < 8; ++r) {
      float v = acc[s][r] + bn;
      float g = 0.5f * v * (1.0f + erff(v * 0.70710678118654752f));  // exact GELU
      out[(size_t)(m0 + mh + r) * HID + n] = (_Float16)g;
    }
  }
}

// FC2 + residual: out = y + act @ w2^T + b2. M=NT, N=96 (6 tiles/wave), K=384 (12 steps)
__global__ __launch_bounds__(256) void fc2_gemm_kernel(
    const _Float16* __restrict__ A, const _Float16* __restrict__ W,
    const float* __restrict__ bias, const float* __restrict__ y, float* __restrict__ out) {
  const int lane = threadIdx.x & 31;
  const int gm   = blockIdx.x * 8 + (threadIdx.x >> 5);
  const int m0   = gm << 4;
  __builtin_prefetch(W, 0, 1);
  v8f acc[6] = {};
  for (int k0 = 0; k0 < HID; k0 += 32) {    // 12 k-steps
    v16h a = frag_a(A + (size_t)m0 * HID + k0, HID, lane);
#pragma unroll
    for (int s = 0; s < 6; ++s) {
      v16h b = frag_b_t(W + (size_t)(16 * s) * HID + k0, HID, lane);
      acc[s] = wmma16(a, b, acc[s]);
    }
  }
  const int mh = (lane >> 4) << 3;
  const int nl = lane & 15;
#pragma unroll
  for (int r = 0; r < 8; ++r) {
    int row = m0 + mh + r;                  // raster token
    int b_ = row / HWSZ, hw = row - b_ * HWSZ;
    size_t base0 = (size_t)b_ * CDIM * HWSZ + hw;
#pragma unroll
    for (int s = 0; s < 6; ++s) {
      int n = 16 * s + nl;                  // channel
      size_t idx = base0 + (size_t)n * HWSZ;
      out[idx] = y[idx] + (acc[s][r] + bias[n]);
    }
  }
}

// ---------------- attention kernel: one block per (window, head) ----------------
__global__ __launch_bounds__(128) void attn_kernel(
    const _Float16* __restrict__ qkv, const float* __restrict__ rel_table,
    _Float16* __restrict__ attn_out) {
  __shared__ _Float16 sQ[64][HD];
  __shared__ _Float16 sK[64][HD];
  __shared__ _Float16 sV[64][HD];
  __shared__ float    sS[64][64];
  __shared__ _Float16 sP[64][64];

  const int bid  = blockIdx.x;
  const int win  = bid / NHEAD, head = bid - win * NHEAD;
  const int tid  = threadIdx.x, lane = tid & 31, wave = tid >> 5;
  const size_t rowbase = (size_t)win * NPW;

#if defined(HAVE_ASYNC_LDS)
  // zero-fill pad rows 49..63 via normal LDS stores
  for (int i = tid; i < 3 * 15 * HD; i += 128) {
    int mat = i / (15 * HD), rem = i - mat * (15 * HD);
    int t = NPW + (rem >> 5), d = rem & 31;
    _Float16* dst = (mat == 0) ? &sQ[t][d] : (mat == 1) ? &sK[t][d] : &sV[t][d];
    *dst = (_Float16)0.f;
  }
  // async direct global->LDS copies, 16B per op (GLOBAL_LOAD_ASYNC_TO_LDS_B128)
  for (int i = tid; i < 3 * NPW * 4; i += 128) {
    int mat = i / (NPW * 4), rem = i - mat * (NPW * 4);
    int t = rem >> 2, ch = (rem & 3) << 3;   // 8-half (16B) chunk
    const _Float16* src = qkv + (rowbase + t) * QKVD + mat * CDIM + head * HD + ch;
    _Float16* dst = (mat == 0) ? &sQ[t][ch] : (mat == 1) ? &sK[t][ch] : &sV[t][ch];
    __builtin_amdgcn_global_load_async_to_lds_b128(
        (AS_GLOBAL v4i*)src, (AS_LDS v4i*)dst, 0, 0);
  }
#if __has_builtin(__builtin_amdgcn_s_wait_asynccnt)
  __builtin_amdgcn_s_wait_asynccnt(0);
#else
  asm volatile("s_wait_asynccnt 0" ::: "memory");
#endif
#else
  // fallback: stage q/k/v through VGPRs (pad rows 49..63 with zero)
  for (int i = tid; i < 64 * HD; i += 128) {
    int t = i >> 5, d = i & 31;
    _Float16 q = (_Float16)0.f, k = (_Float16)0.f, v = (_Float16)0.f;
    if (t < NPW) {
      const _Float16* p = qkv + (rowbase + t) * QKVD + head * HD + d;
      q = p[0]; k = p[CDIM]; v = p[2 * CDIM];
    }
    sQ[t][d] = q; sK[t][d] = k; sV[t][d] = v;
  }
#endif
  __syncthreads();

  // S = q @ k^T : wave handles one 16-row M tile, 4 N tiles, single K step (K=32)
  {
    const int m0 = wave << 4;
    v16h a = frag_a(&sQ[m0][0], HD, lane);   // reused across the 4 N tiles
#pragma unroll
    for (int nt = 0; nt < 4; ++nt) {
      const int n0 = nt << 4;
      v8f acc = {};
      v16h b = frag_b_t(&sK[n0][0], HD, lane);   // k^T: B[k][n] = k[n][kd]
      acc = wmma16(a, b, acc);
      const int n = n0 + (lane & 15), mh = (lane >> 4) << 3;
#pragma unroll
      for (int r = 0; r < 8; ++r) sS[m0 + mh + r][n] = acc[r];
    }
  }
  __syncthreads();

  // softmax per row with relative-position bias + shift mask, -> f16 P
  if (tid < 64) {
    const int t = tid;
    if (t < NPW) {
      const float scale = 0.17677669529663687f;   // 1/sqrt(32)
      const int mi = t / 7, mj = t % 7;
      const int wi = win & 1023, wh = wi >> 5, ww = wi & 31;
      const int gr = wh * 7 + mi, gc = ww * 7 + mj;
      const int rr = gr < 217 ? 0 : (gr < 221 ? 1 : 2);
      const int rc = gc < 217 ? 0 : (gc < 221 ? 1 : 2);
      const int code_m = rr * 3 + rc;
      // pass 1: logits (write back into sS) + max
      float mx = -1e30f;
      for (int j = 0; j < NPW; ++j) {
        int ni = j / 7, nj = j % 7;
        int ridx = (mi - ni + 6) * 13 + (mj - nj + 6);
        float bias = rel_table[ridx * NHEAD + head];
        int gr2 = wh * 7 + ni, gc2 = ww * 7 + nj;
        int rr2 = gr2 < 217 ? 0 : (gr2 < 221 ? 1 : 2);
        int rc2 = gc2 < 217 ? 0 : (gc2 < 221 ? 1 : 2);
        float msk = (code_m == rr2 * 3 + rc2) ? 0.f : -100.f;
        float v = sS[t][j] * scale + bias + msk;
        sS[t][j] = v;
        mx = fmaxf(mx, v);
      }
      // pass 2: exp + sum
      float sum = 0.f;
      for (int j = 0; j < NPW; ++j) {
        float e = __expf(sS[t][j] - mx);
        sS[t][j] = e;
        sum += e;
      }
      float inv = 1.f / sum;
      for (int j = 0; j < NPW; ++j) sP[t][j] = (_Float16)(sS[t][j] * inv);
      for (int j = NPW; j < 64; ++j) sP[t][j] = (_Float16)0.f;
    } else {
      for (int j = 0; j < 64; ++j) sP[t][j] = (_Float16)0.f;
    }
  }
  __syncthreads();

  // O = P @ V : wave -> 16-row M tile, 2 N tiles (hd=32), 2 K steps (64 padded tokens)
  {
    const int m0 = wave << 4;
#pragma unroll
    for (int nt = 0; nt < 2; ++nt) {
      const int n0 = nt << 4;
      v8f acc = {};
#pragma unroll
      for (int ks = 0; ks < 2; ++ks) {
        const int k0 = ks << 5;
        v16h a = frag_a(&sP[m0][k0], 64, lane);
        v16h b = frag_b_k(&sV[k0][n0], HD, lane);  // B[k][n] = v[token k][hd n]
        acc = wmma16(a, b, acc);
      }
      const int n = n0 + (lane & 15), mh = (lane >> 4) << 3;
#pragma unroll
      for (int r = 0; r < 8; ++r) {
        int m = m0 + mh + r;
        if (m < NPW)
          attn_out[(rowbase + m) * CDIM + head * HD + n] = (_Float16)acc[r];
      }
    }
  }
}

// ---------------- host launcher ----------------
extern "C" void kernel_launch(void* const* d_in, const int* in_sizes, int n_in,
                              void* d_out, int out_size, void* d_ws, size_t ws_size,
                              hipStream_t stream) {
  const float* x      = (const float*)d_in[0];
  const float* n1g    = (const float*)d_in[1];
  const float* n1b    = (const float*)d_in[2];
  const float* qkv_w  = (const float*)d_in[3];
  const float* qkv_b  = (const float*)d_in[4];
  const float* rel    = (const float*)d_in[5];
  const float* proj_w = (const float*)d_in[6];
  const float* proj_b = (const float*)d_in[7];
  const float* n2g    = (const float*)d_in[8];
  const float* n2b    = (const float*)d_in[9];
  const float* w1     = (const float*)d_in[10];
  const float* b1     = (const float*)d_in[11];
  const float* w2     = (const float*)d_in[12];
  const float* b2     = (const float*)d_in[13];
  float* out = (float*)d_out;

  // workspace carve-up (256B aligned slabs)
  char* ws = (char*)d_ws;
  size_t off = 0;
  auto take = [&](size_t bytes) -> void* {
    void* p = ws + off;
    off += (bytes + 255) & ~(size_t)255;
    return p;
  };
  _Float16* qkvw16  = (_Float16*)take((size_t)QKVD * CDIM * 2);
  _Float16* projw16 = (_Float16*)take((size_t)CDIM * CDIM * 2);
  _Float16* w116    = (_Float16*)take((size_t)HID * CDIM * 2);
  _Float16* w216    = (_Float16*)take((size_t)CDIM * HID * 2);
  _Float16* xwin16  = (_Float16*)take((size_t)NT * CDIM * 2);
  _Float16* qkv16   = (_Float16*)take((size_t)NT * QKVD * 2);
  _Float16* ao16    = (_Float16*)take((size_t)NT * CDIM * 2);
  float*    ybuf    = (float*)   take((size_t)NT * CDIM * 4);
  _Float16* h16     = (_Float16*)take((size_t)NT * CDIM * 2);
  _Float16* act16   = (_Float16*)take((size_t)NT * HID * 2);

  // 1) weights -> f16
  cvt_f16_kernel<<<(QKVD * CDIM + 255) / 256, 256, 0, stream>>>(qkv_w, qkvw16, QKVD * CDIM);
  cvt_f16_kernel<<<(CDIM * CDIM + 255) / 256, 256, 0, stream>>>(proj_w, projw16, CDIM * CDIM);
  cvt_f16_kernel<<<(HID * CDIM + 255) / 256, 256, 0, stream>>>(w1, w116, HID * CDIM);
  cvt_f16_kernel<<<(CDIM * HID + 255) / 256, 256, 0, stream>>>(w2, w216, CDIM * HID);

  // 2) LN1 + shift + window partition
  ln1_shift_part_kernel<<<NT / 256, 256, 0, stream>>>(x, n1g, n1b, xwin16);

  // 3) QKV GEMM: wave groups = (NT/16) * 3, 8 waves/block
  qkv_gemm_kernel<<<(NT / 16) * 3 / 8, 256, 0, stream>>>(xwin16, qkvw16, qkv_b, qkv16);

  // 4) windowed attention, one block per (window, head)
  attn_kernel<<<NWIN * NHEAD, 128, 0, stream>>>(qkv16, rel, ao16);

  // 5) proj GEMM + window reverse + un-shift + residual: (NT/16) wave groups
  proj_gemm_scatter_kernel<<<(NT / 16) / 8, 256, 0, stream>>>(ao16, projw16, proj_b, x, ybuf);

  // 6) LN2
  ln2_kernel<<<NT / 256, 256, 0, stream>>>(ybuf, n2g, n2b, h16);

  // 7) FC1 + GELU: (NT/16) * 4 wave groups
  fc1_gemm_kernel<<<(NT / 16) * 4 / 8, 256, 0, stream>>>(h16, w116, b1, act16);

  // 8) FC2 + residual -> out: (NT/16) wave groups
  fc2_gemm_kernel<<<(NT / 16) / 8, 256, 0, stream>>>(act16, w216, b2, ybuf, out);
}
